// SSMBlock_81750407512757
// MI455X (gfx1250) — matrix operand order repaired and checked by reading, not compiled, not run on hardware
//
#include <hip/hip_runtime.h>
#include <cstdint>
#include <cstddef>

// ---------------------------------------------------------------------------
// Types for CDNA5 WMMA (gfx1250, wave32)
// ---------------------------------------------------------------------------
typedef __attribute__((ext_vector_type(16))) __bf16 v16bf;
typedef __attribute__((ext_vector_type(8)))  float  v8f;
typedef __attribute__((ext_vector_type(4)))  int    v4i;

union FragBF {
    unsigned int u[8];
    v16bf v;
};

__device__ __forceinline__ unsigned short f2bf(float f) {
    unsigned int u = __float_as_uint(f);
    unsigned int r = (u + 0x7FFFu + ((u >> 16) & 1u)) >> 16;   // round-nearest-even
    return (unsigned short)r;
}
__device__ __forceinline__ float sigmoidf_(float x) { return 1.0f / (1.0f + __expf(-x)); }

// ---------------------------------------------------------------------------
// Async global->LDS (CDNA5 GLOBAL_LOAD_ASYNC_TO_LDS_B128), with fallback.
// Builtin signature: (v4i AS1* src, v4i AS3* dst, imm offset, imm cpol).
// AS(3) pointers are 32-bit; a flat LDS address carries the LDS offset in
// its low 32 bits (ISA aperture rules), so integer casts produce valid
// operands.
// ---------------------------------------------------------------------------
#if __has_builtin(__builtin_amdgcn_global_load_async_to_lds_b128)
#define GEMM_ASYNC 1
#else
#define GEMM_ASYNC 0
#endif

typedef __attribute__((address_space(1))) v4i gas_v4i;
typedef __attribute__((address_space(3))) v4i las_v4i;

__device__ __forceinline__ void copy_b128_g2l(void* lds, const void* g) {
#if GEMM_ASYNC
    __builtin_amdgcn_global_load_async_to_lds_b128(
        (gas_v4i*)(uintptr_t)g,
        (las_v4i*)(unsigned int)(uintptr_t)lds,
        0, 0);
#else
    *(uint4*)lds = *(const uint4*)g;
#endif
}

// Wait until at most `n` of this wave's async ops remain outstanding.
// (Async loads complete in order; each stage issues exactly 4 ops per wave,
// so waiting to <=4 retires the previous tile while the new one is in flight.)
__device__ __forceinline__ void gemm_wait_async4() {
#if GEMM_ASYNC
#if __has_builtin(__builtin_amdgcn_s_wait_asynccnt)
    __builtin_amdgcn_s_wait_asynccnt(4);
#else
    asm volatile("s_wait_asynccnt 0x4" ::: "memory");
#endif
#endif
}
__device__ __forceinline__ void gemm_wait_async0() {
#if GEMM_ASYNC
#if __has_builtin(__builtin_amdgcn_s_wait_asynccnt)
    __builtin_amdgcn_s_wait_asynccnt(0);
#else
    asm volatile("s_wait_asynccnt 0x0" ::: "memory");
#endif
#endif
}

// ---------------------------------------------------------------------------
// f32 -> bf16 cast, 4 elements/thread (all our sizes are multiples of 4).
// ---------------------------------------------------------------------------
__global__ void cast_bf16x4_kernel(const float* __restrict__ src,
                                   unsigned short* __restrict__ dst, int n4) {
    int i = blockIdx.x * blockDim.x + threadIdx.x;
    if (i >= n4) return;
    float4 v = ((const float4*)src)[i];
    uint2 p;
    p.x = (unsigned int)f2bf(v.x) | ((unsigned int)f2bf(v.y) << 16);
    p.y = (unsigned int)f2bf(v.z) | ((unsigned int)f2bf(v.w) << 16);
    ((uint2*)dst)[i] = p;
}

// ---------------------------------------------------------------------------
// Generic bf16 WMMA GEMM:  C[M][N] = A[M][K] * W[N][K]^T   (C in f32)
// Block tile 128(M) x 128(N), K-step 32. 8 waves as 2(M) x 4(N); each wave
// owns a 64x32 macro-tile = 4x2 v_wmma_f32_16x16x32_bf16 per K-step
// (8 WMMAs per 12 ds_load_b128: high matrix-op density).
// M,K multiples of 128/32. N may be ragged (tiles pre-zeroed once;
// redirected loads; guarded stores).
// Staging: double-buffered 16B async DMA chunks straight into LDS, exactly
// 4 async ops per wave per tile so partial ASYNCcnt waits are exact.
// ---------------------------------------------------------------------------
#define TM 128
#define TN 128
#define TK 32
#define LDT 40   // padded row stride (bf16 elems): 80B, keeps 16B chunk alignment

__global__ __launch_bounds__(256) void gemm_bf16_wmma(
    const unsigned short* __restrict__ A,   // [M][K] bf16 bits
    const unsigned short* __restrict__ W,   // [N][K] bf16 bits
    float* __restrict__ C,                  // [M][N] f32
    int M, int N, int K)
{
    __shared__ __align__(16) unsigned short At[2][TM * LDT];
    __shared__ __align__(16) unsigned short Wt[2][TN * LDT];
    __shared__ __align__(16) unsigned short dummy16[8];   // sink for ragged rows

    const int tid  = threadIdx.x;
    const int lane = tid & 31;
    const int wave = tid >> 5;
    const int wm   = (wave & 1) * 64;     // wave M offset in block tile
    const int wn   = (wave >> 1) * 32;    // wave N offset in block tile
    const int blockM = blockIdx.y * TM;
    const int blockN = blockIdx.x * TN;

    const int mrow   = lane & 15;         // M row (A frag) / N col (B frag, C tile)
    const int hi     = lane >> 4;         // lane half
    const int khalfA = hi * 8;            // A-frag K base per documented layout
    const int kbaseB = hi * 16;           // unused; B frag uses khalfA (same K order as A)

    // Ragged-N blocks: pre-zero both Wt buffers once; rows >= N are never
    // rewritten, so fragments read zeros there.
    if (blockN + TN > N) {
        uint4 z = {0u, 0u, 0u, 0u};
        for (int i = tid; i < (2 * TN * LDT * 2) / 16; i += 256)
            ((uint4*)&Wt[0][0])[i] = z;
        __syncthreads();
    }

    // Stage one K-tile into buffer `buf`: exactly 4 async ops per thread/wave.
    auto stage = [&](int buf, int k0) {
        // A: 128 rows x 4 chunks = 512 chunks, 2 per thread
        #pragma unroll
        for (int it = 0; it < 2; ++it) {
            int c = tid + it * 256;
            int m = c >> 2, q = (c & 3) << 3;
            copy_b128_g2l(&At[buf][m * LDT + q],
                          &A[(size_t)(blockM + m) * K + k0 + q]);
        }
        // W: 128 rows x 4 chunks = 512 chunks, 2 per thread.
        // Invalid rows: clamp source, redirect dest to dummy (no branch, so
        // every wave issues the op and per-wave ASYNCcnt stays uniform).
        #pragma unroll
        for (int it = 0; it < 2; ++it) {
            int c = tid + it * 256;
            int n = c >> 2, q = (c & 3) << 3;
            int nrow = blockN + n;
            bool valid = nrow < N;
            const unsigned short* src = &W[(size_t)(valid ? nrow : (N - 1)) * K + k0 + q];
            void* dst = valid ? (void*)&Wt[buf][n * LDT + q] : (void*)&dummy16[0];
            copy_b128_g2l(dst, src);
        }
    };

    v8f acc[4][2] = {};

    stage(0, 0);
    int buf = 0;
    for (int k0 = 0; k0 < K; k0 += TK, buf ^= 1) {
        if (k0 + TK < K) {
            __syncthreads();              // all waves done reading buf^1
            stage(buf ^ 1, k0 + TK);      // prefetch next tile
            gemm_wait_async4();           // current tile's 4 ops retired
        } else {
            gemm_wait_async0();           // last tile: drain everything
        }
        __syncthreads();                  // current tile visible block-wide

        // --- build fragments per documented bf16 VGPR layouts ---
        FragBF a[4], b[2];
        #pragma unroll
        for (int tm = 0; tm < 4; ++tm) {
            #pragma unroll
            for (int j = 0; j < 8; ++j) {
                // A 16x32: VGPR j covers K = (j/4)*16 + hi*8 + 2*(j%4) .. +1
                int kk = ((j >> 2) << 4) + khalfA + ((j & 3) << 1);
                a[tm].u[j] = *(const unsigned int*)&At[buf][(wm + tm * 16 + mrow) * LDT + kk];
            }
        }
        #pragma unroll
        for (int tn = 0; tn < 2; ++tn) {
            #pragma unroll
            for (int j = 0; j < 8; ++j) {
                // B 32x16: VGPR j covers K = (j/4)*16 + hi*8 + 2*(j%4) .. +1; N = lane%16
                int kb = ((j >> 2) << 4) + khalfA + ((j & 3) << 1);
                b[tn].u[j] = *(const unsigned int*)&Wt[buf][(wn + tn * 16 + mrow) * LDT + kb];
            }
        }
        #pragma unroll
        for (int tm = 0; tm < 4; ++tm)
            #pragma unroll
            for (int tn = 0; tn < 2; ++tn)
                acc[tm][tn] = __builtin_amdgcn_wmma_f32_16x16x32_bf16(
                    false, a[tm].v, false, b[tn].v, (short)0, acc[tm][tn], false, false);
    }

    // Store: C/D 16x16 layout -> VGPR r holds M = r + 8*hi, N = lane%16
    #pragma unroll
    for (int tm = 0; tm < 4; ++tm) {
        #pragma unroll
        for (int tn = 0; tn < 2; ++tn) {
            int row0 = blockM + wm + tm * 16 + hi * 8;
            int col  = blockN + wn + tn * 16 + mrow;
            if (col < N) {
                #pragma unroll
                for (int r = 0; r < 8; ++r)
                    C[(size_t)(row0 + r) * N + col] = acc[tm][tn][r];
            }
        }
    }
}

// ---------------------------------------------------------------------------
// Depthwise causal conv (K=4) + bias + SiLU. Writes f32 (for scan) and bf16
// (for the x-projection GEMM).
// ---------------------------------------------------------------------------
__global__ void conv_silu_kernel(const float* __restrict__ xp,   // [B*L][DI]
                                 const float* __restrict__ cw,   // [DI][4]
                                 const float* __restrict__ cb,   // [DI]
                                 float* __restrict__ xa,
                                 unsigned short* __restrict__ xaBf,
                                 int Bsz, int L, int DI)
{
    int idx = blockIdx.x * blockDim.x + threadIdx.x;
    int total = Bsz * L * DI;
    if (idx >= total) return;
    int c = idx % DI;
    int l = (idx / DI) % L;
    int b = idx / (DI * L);
    float acc = cb[c];
    const float* w = &cw[c * 4];
    #pragma unroll
    for (int k = 0; k < 4; ++k) {
        int lsrc = l + k - 3;
        if (lsrc >= 0) acc += w[k] * xp[((size_t)(b * L + lsrc)) * DI + c];
    }
    float v = acc * sigmoidf_(acc);   // silu
    xa[idx]   = v;
    xaBf[idx] = f2bf(v);
}

// ---------------------------------------------------------------------------
// Selective-scan. One thread per (b, d, s): 131072 threads, sequential in L.
// 16-lane shuffle-tree reduces state*C over s. Fuses D*x skip, SiLU(z) gate,
// bf16 output for the final GEMM, and final-state writeout.
// ---------------------------------------------------------------------------
__global__ __launch_bounds__(256) void scan_kernel(
    const float* __restrict__ xa,     // [B*L][DI]
    const float* __restrict__ proj,   // [B*L][33]  (B:0-15, C:16-31, dtr:32)
    const float* __restrict__ zf,     // [B*L][DI]
    const float* __restrict__ ssm0,   // [B][DI][16]
    const float* __restrict__ A_log,  // [DI][16]
    const float* __restrict__ Dp,     // [DI]
    const float* __restrict__ W_dt,   // [DI]
    const float* __restrict__ b_dt,   // [DI]
    unsigned short* __restrict__ yBf, // [B*L][DI]
    float* __restrict__ state_out,    // [B][DI][16]
    int Bsz, int L, int DI)
{
    int gid   = blockIdx.x * blockDim.x + threadIdx.x;
    int per_b = DI * 16;
    int b   = gid / per_b;
    int rem = gid % per_b;
    int d   = rem >> 4;
    int s   = rem & 15;
    if (b >= Bsz) return;

    float Aval  = -__expf(A_log[d * 16 + s]);
    float state = ssm0[(size_t)gid];            // flat [b][d][s] == gid
    float Wdt = W_dt[d], bdt = b_dt[d], Dd = Dp[d];
    const size_t rowBase = (size_t)b * L;

    for (int l = 0; l < L; ++l) {
        size_t r = rowBase + l;
        const float* pr = &proj[r * 33];
        float dtr = pr[32];
        float Bt  = pr[s];
        float Ct  = pr[16 + s];
        float xt  = xa[r * DI + d];
        float va  = dtr * Wdt + bdt;
        float dt  = (va > 20.0f) ? va : log1pf(__expf(va));   // softplus
        state = __expf(dt * Aval) * state + (dt * Bt) * xt;
        float p = state * Ct;
        p += __shfl_xor(p, 1);
        p += __shfl_xor(p, 2);
        p += __shfl_xor(p, 4);
        p += __shfl_xor(p, 8);
        if (s == 0) {
            float y  = p + Dd * xt;
            float zt = zf[r * DI + d];
            yBf[r * DI + d] = f2bf(y * zt * sigmoidf_(zt));
        }
    }
    state_out[(size_t)gid] = state;
}

// ---------------------------------------------------------------------------
// Launch
// ---------------------------------------------------------------------------
extern "C" void kernel_launch(void* const* d_in, const int* in_sizes, int n_in,
                              void* d_out, int out_size, void* d_ws, size_t ws_size,
                              hipStream_t stream)
{
    const int Bsz = 2, L = 2048, DM = 2048, DI = 4096;
    const int M = Bsz * L;   // 4096 token rows

    const float* x     = (const float*)d_in[0];
    const float* ssm0  = (const float*)d_in[1];
    const float* W_in  = (const float*)d_in[2];
    const float* cw    = (const float*)d_in[3];
    const float* cb    = (const float*)d_in[4];
    const float* W_x   = (const float*)d_in[5];
    const float* A_log = (const float*)d_in[6];
    const float* Dp    = (const float*)d_in[7];
    const float* W_dt  = (const float*)d_in[8];
    const float* b_dt  = (const float*)d_in[9];
    const float* W_out = (const float*)d_in[10];

    char* ws = (char*)d_ws;
    size_t o = 0;
    auto alloc = [&](size_t bytes) -> char* {
        char* p = ws + o;
        o += (bytes + 255) & ~(size_t)255;
        return p;
    };
    unsigned short* xBf    = (unsigned short*)alloc((size_t)M * DM * 2);
    unsigned short* WinBf  = (unsigned short*)alloc((size_t)2 * DI * DM * 2);
    unsigned short* WxBf   = (unsigned short*)alloc((size_t)33 * DI * 2);
    unsigned short* WoutBf = (unsigned short*)alloc((size_t)DM * DI * 2);
    float*          xp     = (float*)alloc((size_t)M * DI * 4);
    float*          zf     = (float*)alloc((size_t)M * DI * 4);
    float*          xa     = (float*)alloc((size_t)M * DI * 4);
    unsigned short* xaBf   = (unsigned short*)alloc((size_t)M * DI * 2);
    float*          proj   = (float*)alloc((size_t)M * 33 * 4);
    unsigned short* yBf    = (unsigned short*)alloc((size_t)M * DI * 2);

    auto cgrid4 = [](int n) { return (n / 4 + 255) / 256; };

    // Casts to bf16 (vectorized: all sizes are multiples of 4)
    cast_bf16x4_kernel<<<cgrid4(M * DM),      256, 0, stream>>>(x,     xBf,    (M * DM) / 4);
    cast_bf16x4_kernel<<<cgrid4(2 * DI * DM), 256, 0, stream>>>(W_in,  WinBf,  (2 * DI * DM) / 4);
    cast_bf16x4_kernel<<<cgrid4(33 * DI),     256, 0, stream>>>(W_x,   WxBf,   (33 * DI) / 4);
    cast_bf16x4_kernel<<<cgrid4(DM * DI),     256, 0, stream>>>(W_out, WoutBf, (DM * DI) / 4);

    dim3 blk(256);
    // GEMM1a: xp = x @ W_in[:DI]^T      (M=4096, N=4096, K=2048)
    gemm_bf16_wmma<<<dim3(DI / TN, M / TM), blk, 0, stream>>>(xBf, WinBf, xp, M, DI, DM);
    // GEMM1b: z  = x @ W_in[DI:]^T
    gemm_bf16_wmma<<<dim3(DI / TN, M / TM), blk, 0, stream>>>(xBf, WinBf + (size_t)DI * DM, zf, M, DI, DM);
    // Conv + SiLU
    conv_silu_kernel<<<(M * DI + 255) / 256, 256, 0, stream>>>(xp, cw, cb, xa, xaBf, Bsz, L, DI);
    // GEMM2: proj = xa @ W_x^T          (M=4096, N=33, K=4096)
    gemm_bf16_wmma<<<dim3(1, M / TM), blk, 0, stream>>>(xaBf, WxBf, proj, M, 33, DI);
    // Scan (+ gating, + final state to d_out tail)
    float* out       = (float*)d_out;
    float* state_out = out + (size_t)M * DM;
    scan_kernel<<<(Bsz * DI * 16) / 256, 256, 0, stream>>>(
        xa, proj, zf, ssm0, A_log, Dp, W_dt, b_dt, yBf, state_out, Bsz, L, DI);
    // GEMM3: out = y @ W_out^T          (M=4096, N=2048, K=4096)
    gemm_bf16_wmma<<<dim3(DM / TN, M / TM), blk, 0, stream>>>(yBf, WoutBf, out, M, DM, DI);
}
